// BlockwiseParallel_16879221473609
// MI455X (gfx1250) — matrix operand order repaired and checked by reading, not compiled, not run on hardware
//
#include <hip/hip_runtime.h>
#include <math.h>

// ---------------------------------------------------------------------------
// GPT-J block for MI455X (gfx1250).
// All GEMMs: v_wmma_f32_16x16x32_bf16, double-buffered LDS tiles filled by
// async global->LDS DMA (global_load_async_to_lds_b128 / s_wait_asynccnt).
// ---------------------------------------------------------------------------

typedef __attribute__((ext_vector_type(16))) __bf16          v16bf;
typedef __attribute__((ext_vector_type(8)))  float           v8f;
typedef __attribute__((ext_vector_type(8)))  unsigned short  vu8;  // 16B of bf16 bits

static constexpr int B_  = 2;
static constexpr int S_  = 2048;
static constexpr int D_  = 2048;
static constexpr int H_  = 16;
static constexpr int DH_ = 128;
static constexpr int FF_ = 8192;

union Frag { v16bf v; vu8 h[2]; };

__device__ __forceinline__ unsigned short f2bf(float f) {
  unsigned u = __builtin_bit_cast(unsigned, f);
  u += 0x7FFFu + ((u >> 16) & 1u);          // round-to-nearest-even
  return (unsigned short)(u >> 16);
}

// Async 16B global->LDS copy (per-lane). LDS offset = low 32 bits of the
// generic shared-pointer address (ISA 10.2).
__device__ __forceinline__ void async_b128(void* lds, const void* g) {
  unsigned l = (unsigned)(unsigned long long)lds;
  unsigned long long ga = (unsigned long long)g;
  asm volatile("global_load_async_to_lds_b128 %0, %1, off"
               :: "v"(l), "v"(ga) : "memory");
}
__device__ __forceinline__ void wait_async0() {
  asm volatile("s_wait_asynccnt 0x0" ::: "memory");
}

// ---------------------------------------------------------------------------
// bf16 WMMA GEMM: C[M,N] = A[M,K] x B[N,K]^T   (B always stored N-major)
// Batched via grid.z: off = (z/Hdim)*s?b + (z%Hdim)*s?h for A, B, C.
// Epilogue: +bias[n], optional exact GELU, +resid[idx], stores f32 and/or bf16.
// causal=1 skips tiles entirely above the diagonal.
// Tiling: 128x128x32, 8 waves (4x2), wave = 32x64 = 2x4 WMMA tiles.
// ---------------------------------------------------------------------------
__launch_bounds__(256)
__global__ void gemm_bf16_wmma(const unsigned short* __restrict__ A,
                               const unsigned short* __restrict__ Bm,
                               float* __restrict__ Cf,
                               unsigned short* __restrict__ Cbf,
                               const float* __restrict__ bias,
                               const float* __restrict__ resid,
                               int M, int N, int K,
                               int lda, int ldb, int ldc,
                               long long sAb, long long sAh,
                               long long sBb, long long sBh,
                               long long sCb, long long sCh,
                               int Hdim, int act, int causal)
{
  const int bx = blockIdx.x, by = blockIdx.y, z = blockIdx.z;
  if (causal && (by * 128 + 127) < (bx * 128)) return;   // fully masked tile

  const long long offA = (long long)(z / Hdim) * sAb + (long long)(z % Hdim) * sAh;
  const long long offB = (long long)(z / Hdim) * sBb + (long long)(z % Hdim) * sBh;
  const long long offC = (long long)(z / Hdim) * sCb + (long long)(z % Hdim) * sCh;

  const unsigned short* Ag = A + offA + (long long)by * 128 * lda;
  const unsigned short* Bg = Bm + offB + (long long)bx * 128 * ldb;

  // double-buffered tiles, pitch 40 u16 = 80B (16B multiple)
  __shared__ __align__(16) unsigned short As[2][128][40];
  __shared__ __align__(16) unsigned short Bs[2][128][40];

  const int tid  = threadIdx.x;
  const int wave = tid >> 5, lane = tid & 31;
  const int wm   = wave & 3;          // 4 waves along M
  const int wn   = wave >> 2;         // 2 waves along N
  const int half = lane >> 4;
  const int mn   = lane & 15;

  const int srow = tid >> 2;          // 0..63  (staging row)
  const int scol = (tid & 3) * 8;     // 0,8,16,24 (staging k-chunk)

  auto stage = [&](int buf, int k0) {
    async_b128(&As[buf][srow][scol],      Ag + (long long)srow * lda + k0 + scol);
    async_b128(&As[buf][srow + 64][scol], Ag + (long long)(srow + 64) * lda + k0 + scol);
    async_b128(&Bs[buf][srow][scol],      Bg + (long long)srow * ldb + k0 + scol);
    async_b128(&Bs[buf][srow + 64][scol], Bg + (long long)(srow + 64) * ldb + k0 + scol);
  };

  const v8f zero8 = {0.f, 0.f, 0.f, 0.f, 0.f, 0.f, 0.f, 0.f};
  v8f acc[2][4];
#pragma unroll
  for (int mi = 0; mi < 2; ++mi)
#pragma unroll
    for (int ni = 0; ni < 4; ++ni) acc[mi][ni] = zero8;

  // prologue: fill buffer 0
  stage(0, 0);
  wait_async0();
  __syncthreads();

  int buf = 0;
  for (int k0 = 0; k0 < K; k0 += 32) {
    if (k0 + 32 < K) stage(buf ^ 1, k0 + 32);   // DMA next tile during compute

    Frag a[2], b[4];
#pragma unroll
    for (int mi = 0; mi < 2; ++mi) {
      int r = wm * 32 + mi * 16 + mn;
      a[mi].h[0] = *(const vu8*)&As[buf][r][half * 8];       // K 0..7 / 8..15
      a[mi].h[1] = *(const vu8*)&As[buf][r][16 + half * 8];  // K 16..23 / 24..31
    }
#pragma unroll
    for (int ni = 0; ni < 4; ++ni) {
      int r = wn * 64 + ni * 16 + mn;
      b[ni].h[0] = *(const vu8*)&Bs[buf][r][half * 8];       // K 0..7 / 8..15
      b[ni].h[1] = *(const vu8*)&Bs[buf][r][16 + half * 8];  // K 16..23 / 24..31
    }
#pragma unroll
    for (int mi = 0; mi < 2; ++mi)
#pragma unroll
      for (int ni = 0; ni < 4; ++ni)
        acc[mi][ni] = __builtin_amdgcn_wmma_f32_16x16x32_bf16(
            false, a[mi].v, false, b[ni].v, (short)0, acc[mi][ni], false, false);

    wait_async0();        // next-tile DMA complete (this wave)
    __syncthreads();      // all waves done reading `buf` / writing `buf^1`
    buf ^= 1;
  }

  // ---- epilogue: bias / gelu / residual / dual store ---------------------
#pragma unroll
  for (int mi = 0; mi < 2; ++mi) {
#pragma unroll
    for (int r = 0; r < 8; ++r) {
      int gRow = by * 128 + wm * 32 + mi * 16 + half * 8 + r;
#pragma unroll
      for (int ni = 0; ni < 4; ++ni) {
        int gCol = bx * 128 + wn * 64 + ni * 16 + mn;
        float v = acc[mi][ni][r];
        if (bias) v += bias[gCol];
        if (act == 1) v = 0.5f * v * (1.0f + erff(v * 0.70710678118654752f));
        long long idx = offC + (long long)gRow * ldc + gCol;
        if (resid) v += resid[idx];
        if (Cf)  Cf[idx]  = v;
        if (Cbf) Cbf[idx] = f2bf(v);
      }
    }
  }
}

// ---------------------------------------------------------------------------
// Tiled transpose + f32->bf16 convert: dst[c][r] = src[r][c].
// src: [R x Cc] f32 (row stride lds_), dst: [Cc x R] bf16 (row stride ldd_).
// Batched via grid.z like the GEMM.
// ---------------------------------------------------------------------------
__launch_bounds__(256)
__global__ void transpose_cvt_kernel(const float* __restrict__ src,
                                     unsigned short* __restrict__ dst,
                                     int lds_, int ldd_,
                                     long long sSb, long long sSh,
                                     long long sDb, long long sDh, int Hdim)
{
  __shared__ float t[32][33];
  const int z = blockIdx.z;
  const long long so = (long long)(z / Hdim) * sSb + (long long)(z % Hdim) * sSh;
  const long long dof = (long long)(z / Hdim) * sDb + (long long)(z % Hdim) * sDh;
  const int tx = threadIdx.x & 31, ty = threadIdx.x >> 5;   // 32 x 8

#pragma unroll
  for (int i = 0; i < 32; i += 8) {
    int r = blockIdx.y * 32 + ty + i;
    int c = blockIdx.x * 32 + tx;
    t[ty + i][tx] = src[so + (long long)r * lds_ + c];
  }
  __syncthreads();
#pragma unroll
  for (int i = 0; i < 32; i += 8) {
    int r = blockIdx.x * 32 + ty + i;      // dst row = src col
    int c = blockIdx.y * 32 + tx;          // dst col = src row
    dst[dof + (long long)r * ldd_ + c] = f2bf(t[tx][ty + i]);
  }
}

// ---------------------------------------------------------------------------
// LayerNorm over D=2048 (block per row), optional x+xadd, f32/bf16 outputs.
// ---------------------------------------------------------------------------
__launch_bounds__(256)
__global__ void ln_kernel(const float* __restrict__ x,
                          const float* __restrict__ xadd,
                          const float* __restrict__ g,
                          const float* __restrict__ bsh,
                          float* __restrict__ of,
                          unsigned short* __restrict__ ob)
{
  __shared__ float red[256];
  const long long row = blockIdx.x;
  const float* xr = x + row * D_;
  const float* ar = xadd ? xadd + row * D_ : nullptr;
  const int tid = threadIdx.x;

  float vals[8], s = 0.f;
#pragma unroll
  for (int i = 0; i < 8; ++i) {
    int c = tid + i * 256;
    float v = xr[c];
    if (ar) v += ar[c];
    vals[i] = v; s += v;
  }
  red[tid] = s; __syncthreads();
  for (int st = 128; st > 0; st >>= 1) {
    if (tid < st) red[tid] += red[tid + st];
    __syncthreads();
  }
  float mean = red[0] * (1.0f / D_); __syncthreads();

  float s2 = 0.f;
#pragma unroll
  for (int i = 0; i < 8; ++i) { float d = vals[i] - mean; s2 += d * d; }
  red[tid] = s2; __syncthreads();
  for (int st = 128; st > 0; st >>= 1) {
    if (tid < st) red[tid] += red[tid + st];
    __syncthreads();
  }
  float rstd = rsqrtf(red[0] * (1.0f / D_) + 1e-5f);

#pragma unroll
  for (int i = 0; i < 8; ++i) {
    int c = tid + i * 256;
    float y = (vals[i] - mean) * rstd * g[c] + bsh[c];
    if (of) of[row * D_ + c] = y;
    if (ob) ob[row * D_ + c] = f2bf(y);
  }
}

// ---------------------------------------------------------------------------
// GPT-J interleaved RoPE on q,k (f32 in) -> bf16 out. One thread per pair.
// ---------------------------------------------------------------------------
__launch_bounds__(256)
__global__ void rope_kernel(const float* __restrict__ qf,
                            const float* __restrict__ kf,
                            const int* __restrict__ pos,
                            unsigned short* __restrict__ qb,
                            unsigned short* __restrict__ kb)
{
  const long long total = (long long)B_ * S_ * H_ * (DH_ / 2);
  long long idx = (long long)blockIdx.x * blockDim.x + threadIdx.x;
  if (idx >= total) return;
  int i = (int)(idx % (DH_ / 2));
  long long t = idx / (DH_ / 2);
  int h = (int)(t % H_); t /= H_;
  int s = (int)(t % S_);
  int b = (int)(t / S_);

  int p = pos[b * S_ + s];
  float freq = __powf(10000.0f, -(2.0f * i) / (float)DH_);
  float sn, cs;
  __sincosf((float)p * freq, &sn, &cs);

  long long base = (((long long)b * S_ + s) * D_) + (long long)h * DH_ + 2 * i;
  float q0 = qf[base], q1 = qf[base + 1];
  qb[base]     = f2bf(q0 * cs - q1 * sn);
  qb[base + 1] = f2bf(q1 * cs + q0 * sn);
  float k0 = kf[base], k1 = kf[base + 1];
  kb[base]     = f2bf(k0 * cs - k1 * sn);
  kb[base + 1] = f2bf(k1 * cs + k0 * sn);
}

// ---------------------------------------------------------------------------
// Causal softmax over one score row (S=2048): in-place f32 + bf16 copy.
// Masked positions (k>q) written as exact 0 (== softmax with -1e9 mask).
// ---------------------------------------------------------------------------
__launch_bounds__(256)
__global__ void softmax_causal_kernel(float* __restrict__ sc,
                                      unsigned short* __restrict__ wb)
{
  __shared__ float red[256];
  const long long row = blockIdx.x;          // [0, B*H*S)
  const int q = (int)(row % S_);
  float* r = sc + row * (long long)S_;
  unsigned short* o = wb + row * (long long)S_;
  const int tid = threadIdx.x;

  float mx = -3.4e38f;
  for (int k = tid; k <= q; k += 256) mx = fmaxf(mx, r[k]);
  red[tid] = mx; __syncthreads();
  for (int st = 128; st > 0; st >>= 1) {
    if (tid < st) red[tid] = fmaxf(red[tid], red[tid + st]);
    __syncthreads();
  }
  mx = red[0]; __syncthreads();

  float sum = 0.f;
  for (int k = tid; k <= q; k += 256) sum += expf(r[k] - mx);
  red[tid] = sum; __syncthreads();
  for (int st = 128; st > 0; st >>= 1) {
    if (tid < st) red[tid] += red[tid + st];
    __syncthreads();
  }
  float inv = 1.0f / red[0];

  for (int k = tid; k < S_; k += 256) {
    float w = (k <= q) ? expf(r[k] - mx) * inv : 0.0f;
    r[k] = w;
    o[k] = f2bf(w);
  }
}

// ---------------------------------------------------------------------------
extern "C" void kernel_launch(void* const* d_in, const int* in_sizes, int n_in,
                              void* d_out, int out_size, void* d_ws, size_t ws_size,
                              hipStream_t stream)
{
  (void)in_sizes; (void)n_in; (void)out_size; (void)ws_size;

  const float* hidden  = (const float*)d_in[0];
  const int*   pos     = (const int*)  d_in[1];
  const float* wq      = (const float*)d_in[2];
  const float* wk      = (const float*)d_in[3];
  const float* wv      = (const float*)d_in[4];
  const float* wo      = (const float*)d_in[5];
  const float* ln1_g   = (const float*)d_in[6];
  const float* ln1_b   = (const float*)d_in[7];
  const float* ln2_g   = (const float*)d_in[8];
  const float* ln2_b   = (const float*)d_in[9];
  const float* fcin_w  = (const float*)d_in[10];
  const float* fcin_b  = (const float*)d_in[11];
  const float* fcout_w = (const float*)d_in[12];
  const float* fcout_b = (const float*)d_in[13];

  float* out   = (float*)d_out;                         // [B,S,D] final
  float* attnW = out + (long long)B_ * S_ * D_;         // [B,H,S,S] weights

  const long long nBSD  = (long long)B_ * S_ * D_;
  const long long nDD   = (long long)D_ * D_;
  const long long nDFF  = (long long)D_ * FF_;
  const long long nAttn = (long long)B_ * H_ * S_ * S_;
  const long long nBSFF = (long long)B_ * S_ * FF_;

  // ---- workspace partition ---------------------------------------------
  char* ws = (char*)d_ws;
  size_t off = 0;
  auto alloc = [&](size_t bytes) -> void* {
    void* p = ws + off;
    off += (bytes + 255) & ~(size_t)255;
    return p;
  };
  unsigned short* hln_bf   = (unsigned short*)alloc(nBSD * 2);
  unsigned short* wqT      = (unsigned short*)alloc(nDD * 2);   // [N,K]
  unsigned short* wkT      = (unsigned short*)alloc(nDD * 2);
  unsigned short* wvT      = (unsigned short*)alloc(nDD * 2);
  unsigned short* woT      = (unsigned short*)alloc(nDD * 2);
  unsigned short* fcinT    = (unsigned short*)alloc(nDFF * 2);  // [FF,D]
  unsigned short* fcoutT   = (unsigned short*)alloc(nDFF * 2);  // [D,FF]
  float*          q_f      = (float*)        alloc(nBSD * 4);
  float*          k_f      = (float*)        alloc(nBSD * 4);
  float*          v_f      = (float*)        alloc(nBSD * 4);
  unsigned short* q_bf     = (unsigned short*)alloc(nBSD * 2);
  unsigned short* k_bf     = (unsigned short*)alloc(nBSD * 2);
  unsigned short* vT_bf    = (unsigned short*)alloc(nBSD * 2);  // [B,H,DH,S]
  unsigned short* attn_bf  = (unsigned short*)alloc(nAttn * 2);
  unsigned short* ctx_bf   = (unsigned short*)alloc(nBSD * 2);
  float*          proj_f   = (float*)        alloc(nBSD * 4);
  float*          h2_f     = (float*)        alloc(nBSD * 4);
  unsigned short* h2_bf    = (unsigned short*)alloc(nBSD * 2);
  unsigned short* mid_bf   = (unsigned short*)alloc(nBSFF * 2);

  const int M = B_ * S_;   // 4096
  const long long z0 = 0;

  // ---- 1. LN1 -> bf16 ---------------------------------------------------
  ln_kernel<<<M, 256, 0, stream>>>(hidden, nullptr, ln1_g, ln1_b, nullptr, hln_bf);

  // ---- 2. weight transpose+convert to [N,K] bf16 ------------------------
  {
    dim3 gDD(D_ / 32, D_ / 32, 1);
    transpose_cvt_kernel<<<gDD, 256, 0, stream>>>(wq, wqT, D_, D_, z0, z0, z0, z0, 1);
    transpose_cvt_kernel<<<gDD, 256, 0, stream>>>(wk, wkT, D_, D_, z0, z0, z0, z0, 1);
    transpose_cvt_kernel<<<gDD, 256, 0, stream>>>(wv, wvT, D_, D_, z0, z0, z0, z0, 1);
    transpose_cvt_kernel<<<gDD, 256, 0, stream>>>(wo, woT, D_, D_, z0, z0, z0, z0, 1);
    dim3 gIn(FF_ / 32, D_ / 32, 1);   // src [D,FF] -> dst [FF,D]
    transpose_cvt_kernel<<<gIn, 256, 0, stream>>>(fcin_w, fcinT, FF_, D_, z0, z0, z0, z0, 1);
    dim3 gOut(D_ / 32, FF_ / 32, 1);  // src [FF,D] -> dst [D,FF]
    transpose_cvt_kernel<<<gOut, 256, 0, stream>>>(fcout_w, fcoutT, D_, FF_, z0, z0, z0, z0, 1);
  }

  // ---- 3. Q, K, V projections ------------------------------------------
  dim3 gQKV(D_ / 128, M / 128, 1);
  gemm_bf16_wmma<<<gQKV, 256, 0, stream>>>(hln_bf, wqT, q_f, nullptr, nullptr,
      nullptr, M, D_, D_, D_, D_, D_, z0, z0, z0, z0, z0, z0, 1, 0, 0);
  gemm_bf16_wmma<<<gQKV, 256, 0, stream>>>(hln_bf, wkT, k_f, nullptr, nullptr,
      nullptr, M, D_, D_, D_, D_, D_, z0, z0, z0, z0, z0, z0, 1, 0, 0);
  gemm_bf16_wmma<<<gQKV, 256, 0, stream>>>(hln_bf, wvT, v_f, nullptr, nullptr,
      nullptr, M, D_, D_, D_, D_, D_, z0, z0, z0, z0, z0, z0, 1, 0, 0);

  // ---- 4. RoPE (q,k) + V transpose to [B,H,DH,S] bf16 ------------------
  {
    long long pairs = (long long)B_ * S_ * H_ * (DH_ / 2);
    rope_kernel<<<(int)((pairs + 255) / 256), 256, 0, stream>>>(q_f, k_f, pos, q_bf, k_bf);
    dim3 gV(DH_ / 32, S_ / 32, B_ * H_);   // per (b,h): [S,DH] -> [DH,S]
    transpose_cvt_kernel<<<gV, 256, 0, stream>>>(v_f, vT_bf, D_, S_,
        (long long)S_ * D_, (long long)DH_,
        (long long)H_ * DH_ * S_, (long long)DH_ * S_, H_);
  }

  // ---- 5. scores = Q @ K^T (K already [N,K] per head), causal skip -----
  {
    dim3 g(S_ / 128, S_ / 128, B_ * H_);
    gemm_bf16_wmma<<<g, 256, 0, stream>>>(q_bf, k_bf, attnW, nullptr, nullptr,
        nullptr, S_, S_, DH_, D_, D_, S_,
        (long long)S_ * D_, (long long)DH_,
        (long long)S_ * D_, (long long)DH_,
        (long long)H_ * S_ * S_, (long long)S_ * S_,
        H_, 0, /*causal=*/1);
  }

  // ---- 6. causal softmax (f32 in-place + bf16 copy) --------------------
  softmax_causal_kernel<<<B_ * H_ * S_, 256, 0, stream>>>(attnW, attn_bf);

  // ---- 7. ctx = attn_weights @ V  (B = vT [DH,S] per head) -------------
  {
    dim3 g(DH_ / 128, S_ / 128, B_ * H_);
    gemm_bf16_wmma<<<g, 256, 0, stream>>>(attn_bf, vT_bf, nullptr, ctx_bf, nullptr,
        nullptr, S_, DH_, S_, S_, S_, D_,
        (long long)H_ * S_ * S_, (long long)S_ * S_,
        (long long)H_ * DH_ * S_, (long long)DH_ * S_,
        (long long)S_ * D_, (long long)DH_,
        H_, 0, 0);
  }

  // ---- 8. attention output projection ----------------------------------
  gemm_bf16_wmma<<<gQKV, 256, 0, stream>>>(ctx_bf, woT, proj_f, nullptr, nullptr,
      nullptr, M, D_, D_, D_, D_, D_, z0, z0, z0, z0, z0, z0, 1, 0, 0);

  // ---- 9. h2 = LN2(hidden + attn_proj) ---------------------------------
  ln_kernel<<<M, 256, 0, stream>>>(hidden, proj_f, ln2_g, ln2_b, h2_f, h2_bf);

  // ---- 10. FFN in: gelu(h2 @ fc_in + b) -> bf16 ------------------------
  {
    dim3 g(FF_ / 128, M / 128, 1);
    gemm_bf16_wmma<<<g, 256, 0, stream>>>(h2_bf, fcinT, nullptr, mid_bf, fcin_b,
        nullptr, M, FF_, D_, D_, D_, FF_, z0, z0, z0, z0, z0, z0,
        1, /*gelu=*/1, 0);
  }

  // ---- 11. FFN out + bias + residual(h2) -> d_out ----------------------
  gemm_bf16_wmma<<<gQKV, 256, 0, stream>>>(mid_bf, fcoutT, out, nullptr, fcout_b,
      h2_f, M, D_, FF_, FF_, FF_, D_, z0, z0, z0, z0, z0, z0, 1, 0, 0);
}
